// GATEncoder_31078383354434
// MI455X (gfx1250) — matrix operand order repaired and checked by reading, not compiled, not run on hardware
//
#include <hip/hip_runtime.h>
#include <hip/hip_bf16.h>

typedef __attribute__((ext_vector_type(2)))  float  v2f;
typedef __attribute__((ext_vector_type(8)))  float  v8f;
typedef __attribute__((ext_vector_type(16))) __bf16 v16bf;

// ---------------------------------------------------------------------------
// GEMM: C[M,Nc] = A[M,K] @ B[K,Nc].  One wave (32 lanes) per 16x16 C tile.
// blockDim.x = 32 * (Nc/16); gridDim.x = M/16.  M % 16 == 0, K % 32 == 0.
// ---------------------------------------------------------------------------
__global__ __launch_bounds__(256) void gemm_wmma_kernel(
    const float* __restrict__ A, const float* __restrict__ B,
    float* __restrict__ C, int M, int K, int Nc) {
  const int wave  = threadIdx.x >> 5;
  const int lane  = threadIdx.x & 31;
  const int half  = lane >> 4;      // 0: lanes 0-15, 1: lanes 16-31
  const int l15   = lane & 15;
  const int row0  = blockIdx.x * 16;
  const int col0  = wave * 16;
  if (row0 >= M || col0 >= Nc) return;

  v8f acc = {};
  const float* Arow = A + (size_t)(row0 + l15) * K;

#if defined(__gfx1250__) && __has_builtin(__builtin_amdgcn_wmma_f32_16x16x4_f32)
  // V_WMMA_F32_16X16X4_F32: A 16x4 (lanes0-15: K=k,k+1; lanes16-31: K=k+2,k+3)
  for (int k = 0; k < K; k += 4) {
    v2f a, b;
    const int ka = k + 2 * half;
    a[0] = Arow[ka + 0];
    a[1] = Arow[ka + 1];
    b[0] = B[(size_t)(ka + 0) * Nc + col0 + l15];
    b[1] = B[(size_t)(ka + 1) * Nc + col0 + l15];
    acc = __builtin_amdgcn_wmma_f32_16x16x4_f32(
        false, a, false, b, (short)0, acc, false, false);
  }
#elif defined(__gfx1250__)
  // Fallback: bf16 WMMA 16x16x32 (codegen-confirmed builtin).
  for (int k = 0; k < K; k += 32) {
    v16bf a, b;
#pragma unroll
    for (int i = 0; i < 16; ++i) {
      const int ka = k + (half ? 8 : 0) + (i & 7) + ((i >> 3) * 16);
      a[i] = (__bf16)Arow[ka];
      const int kb = k + (half ? 8 : 0) + (i & 7) + ((i >> 3) * 16);
      b[i] = (__bf16)B[(size_t)kb * Nc + col0 + l15];
    }
    acc = __builtin_amdgcn_wmma_f32_16x16x32_bf16(
        false, a, false, b, (short)0, acc, false, false);
  }
#else
  // Host-parse-only fallback (never executed on device): naive accumulate.
  for (int k = 0; k < K; ++k) {
    const float a = Arow[k];
    const float b = B[(size_t)k * Nc + col0 + l15];
#pragma unroll
    for (int r = 0; r < 8; ++r) acc[r] += a * b;
  }
#endif

  // C/D layout: VGPR r -> row (r + 8*half), column = col0 + l15
#pragma unroll
  for (int r = 0; r < 8; ++r) {
    C[(size_t)(row0 + r + 8 * half) * Nc + col0 + l15] = acc[r];
  }
}

// ---------------------------------------------------------------------------
// Per-node attention halves: als[n,h] = <h_row, a_src>, ald likewise.
// One wave per node; channels = heads*32, lane covers one channel per head.
// ---------------------------------------------------------------------------
__global__ __launch_bounds__(256) void attn_coef_kernel(
    const float* __restrict__ h, const float* __restrict__ a_src,
    const float* __restrict__ a_dst, float* __restrict__ als,
    float* __restrict__ ald, int n, int heads) {
  const int wave = threadIdx.x >> 5;
  const int lane = threadIdx.x & 31;
  const int node = blockIdx.x * (blockDim.x >> 5) + wave;
  if (node >= n) return;
  const float* row = h + (size_t)node * heads * 32;
  for (int hd = 0; hd < heads; ++hd) {
    const float v = row[hd * 32 + lane];
    float s = v * a_src[hd * 32 + lane];
    float d = v * a_dst[hd * 32 + lane];
#pragma unroll
    for (int off = 16; off > 0; off >>= 1) {
      s += __shfl_xor(s, off, 32);
      d += __shfl_xor(d, off, 32);
    }
    if (lane == 0) {
      als[node * heads + hd] = s;
      ald[node * heads + hd] = d;
    }
  }
}

__global__ void fill_kernel(float* __restrict__ p, float v, int n) {
  const int i = blockIdx.x * blockDim.x + threadIdx.x;
  if (i < n) p[i] = v;
}

__global__ void bias_init_kernel(float* __restrict__ out,
                                 const float* __restrict__ bias, int n, int c) {
  const int i = blockIdx.x * blockDim.x + threadIdx.x;
  if (i < n * c) out[i] = bias[i % c];
}

__device__ inline void atomicMaxF(float* addr, float val) {
  int* ia = reinterpret_cast<int*>(addr);
  int old = *ia;
  while (__int_as_float(old) < val) {
    const int assumed = old;
    old = atomicCAS(ia, assumed, __float_as_int(val));
    if (old == assumed) break;
  }
}

// e = leaky_relu(als[src] + ald[dst]); segment-max over dst.
__global__ void edge_logit_max_kernel(
    const int* __restrict__ ei, const float* __restrict__ als,
    const float* __restrict__ ald, float* __restrict__ elog,
    float* __restrict__ emax, int Eorig, int Etot, int heads) {
  const int t = blockIdx.x * blockDim.x + threadIdx.x;
  if (t >= Etot * heads) return;
  const int e = t / heads, hd = t - e * heads;
  int s, d;
  if (e < Eorig) { s = ei[e]; d = ei[Eorig + e]; } else { s = d = e - Eorig; }
  float v = als[s * heads + hd] + ald[d * heads + hd];
  v = v > 0.0f ? v : 0.2f * v;
  elog[t] = v;
  atomicMaxF(&emax[d * heads + hd], v);
}

// e_exp = exp(e - emax[dst]); denom[dst] += e_exp (stored in place of elog).
__global__ void edge_expsum_kernel(
    const int* __restrict__ ei, float* __restrict__ elog,
    const float* __restrict__ emax, float* __restrict__ denom,
    int Eorig, int Etot, int heads) {
  const int t = blockIdx.x * blockDim.x + threadIdx.x;
  if (t >= Etot * heads) return;
  const int e = t / heads, hd = t - e * heads;
  int d;
  if (e < Eorig) { d = ei[Eorig + e]; } else { d = e - Eorig; }
  const float ex = __expf(elog[t] - emax[d * heads + hd]);
  elog[t] = ex;
  atomicAdd(&denom[d * heads + hd], ex);
}

// agg[dst] += h[src] * alpha.  One wave per edge; lane = channel within head.
__global__ __launch_bounds__(256) void edge_aggregate_kernel(
    const int* __restrict__ ei, const float* __restrict__ h,
    const float* __restrict__ eexp, const float* __restrict__ denom,
    float* __restrict__ agg, int Eorig, int Etot, int heads) {
  const int wave = threadIdx.x >> 5;
  const int lane = threadIdx.x & 31;
  const int e = blockIdx.x * (blockDim.x >> 5) + wave;
  if (e >= Etot) return;
  int s, d;
  if (e < Eorig) { s = ei[e]; d = ei[Eorig + e]; } else { s = d = e - Eorig; }
  const float* hrow = h + (size_t)s * heads * 32;
  float* arow = agg + (size_t)d * heads * 32;
  for (int hd = 0; hd < heads; ++hd) {
    const float alpha = eexp[e * heads + hd] / (denom[d * heads + hd] + 1e-16f);
    atomicAdd(&arow[hd * 32 + lane], hrow[hd * 32 + lane] * alpha);
  }
}

// out = elu(agg + bias)
__global__ void elu_bias_kernel(const float* __restrict__ agg,
                                const float* __restrict__ bias,
                                float* __restrict__ out, int total, int hc) {
  const int i = blockIdx.x * blockDim.x + threadIdx.x;
  if (i >= total) return;
  const float v = agg[i] + bias[i % hc];
  out[i] = v > 0.0f ? v : (__expf(v) - 1.0f);
}

// ---------------------------------------------------------------------------
extern "C" void kernel_launch(void* const* d_in, const int* in_sizes, int n_in,
                              void* d_out, int out_size, void* d_ws,
                              size_t ws_size, hipStream_t stream) {
  (void)in_sizes; (void)n_in; (void)out_size; (void)ws_size;
  const float* x      = (const float*)d_in[0];
  const int*   ei     = (const int*)d_in[1];   // [2,E] flat: src row 0, dst row 1
  const float* W1     = (const float*)d_in[2];
  const float* a_src1 = (const float*)d_in[3];
  const float* a_dst1 = (const float*)d_in[4];
  const float* b1     = (const float*)d_in[5];
  const float* W2     = (const float*)d_in[6];
  const float* a_src2 = (const float*)d_in[7];
  const float* a_dst2 = (const float*)d_in[8];
  const float* b2     = (const float*)d_in[9];
  float* out = (float*)d_out;

  constexpr int N = 50000, E = 800000, ET = E + N;
  constexpr int H1 = 4, C = 32, HC1 = H1 * C;  // 128

  float* ws    = (float*)d_ws;
  float* h1    = ws; ws += (size_t)N * HC1;
  float* agg1  = ws; ws += (size_t)N * HC1;
  float* h2    = ws; ws += (size_t)N * C;
  float* e1    = ws; ws += (size_t)ET * H1;
  float* e2    = ws; ws += (size_t)ET;
  float* als1  = ws; ws += N * H1;
  float* ald1  = ws; ws += N * H1;
  float* emax1 = ws; ws += N * H1;
  float* den1  = ws; ws += N * H1;
  float* als2  = ws; ws += N;
  float* ald2  = ws; ws += N;
  float* emax2 = ws; ws += N;
  float* den2  = ws; ws += N;

  const int T = 256;
  auto cdiv = [](int a, int b) { return (a + b - 1) / b; };

  // ----- Layer 1 -----
  gemm_wmma_kernel<<<N / 16, 32 * (HC1 / 16), 0, stream>>>(x, W1, h1, N, HC1, HC1);
  attn_coef_kernel<<<cdiv(N, 8), T, 0, stream>>>(h1, a_src1, a_dst1, als1, ald1, N, H1);
  fill_kernel<<<cdiv(N * H1, T), T, 0, stream>>>(emax1, -3.0e38f, N * H1);
  fill_kernel<<<cdiv(N * H1, T), T, 0, stream>>>(den1, 0.0f, N * H1);
  fill_kernel<<<cdiv(N * HC1, T), T, 0, stream>>>(agg1, 0.0f, N * HC1);
  edge_logit_max_kernel<<<cdiv(ET * H1, T), T, 0, stream>>>(ei, als1, ald1, e1, emax1, E, ET, H1);
  edge_expsum_kernel<<<cdiv(ET * H1, T), T, 0, stream>>>(ei, e1, emax1, den1, E, ET, H1);
  edge_aggregate_kernel<<<cdiv(ET, 8), T, 0, stream>>>(ei, h1, e1, den1, agg1, E, ET, H1);
  elu_bias_kernel<<<cdiv(N * HC1, T), T, 0, stream>>>(agg1, b1, agg1, N * HC1, HC1);

  // ----- Layer 2 -----
  gemm_wmma_kernel<<<N / 16, 32 * (C / 16), 0, stream>>>(agg1, W2, h2, N, HC1, C);
  attn_coef_kernel<<<cdiv(N, 8), T, 0, stream>>>(h2, a_src2, a_dst2, als2, ald2, N, 1);
  fill_kernel<<<cdiv(N, T), T, 0, stream>>>(emax2, -3.0e38f, N);
  fill_kernel<<<cdiv(N, T), T, 0, stream>>>(den2, 0.0f, N);
  bias_init_kernel<<<cdiv(N * C, T), T, 0, stream>>>(out, b2, N, C);
  edge_logit_max_kernel<<<cdiv(ET, T), T, 0, stream>>>(ei, als2, ald2, e2, emax2, E, ET, 1);
  edge_expsum_kernel<<<cdiv(ET, T), T, 0, stream>>>(ei, e2, emax2, den2, E, ET, 1);
  edge_aggregate_kernel<<<cdiv(ET, 8), T, 0, stream>>>(ei, h2, e2, den2, out, E, ET, 1);
}
